// Attention_two_encoder_72559177499310
// MI455X (gfx1250) — matrix operand order repaired and checked by reading, not compiled, not run on hardware
//
#include <hip/hip_runtime.h>
#include <cstdint>
#include <cstddef>

// ---------------------------------------------------------------------------
// Types matching the gfx1250 WMMA builtin signatures
// ---------------------------------------------------------------------------
typedef __bf16 v4bf  __attribute__((ext_vector_type(4)));
typedef __bf16 v8bf  __attribute__((ext_vector_type(8)));
typedef __bf16 v16bf __attribute__((ext_vector_type(16)));
typedef float  v4f   __attribute__((ext_vector_type(4)));
typedef float  v8f   __attribute__((ext_vector_type(8)));

union Frag16 { v16bf v; v8bf h[2]; };

__device__ __forceinline__ v8f zero8() {
  v8f z; 
#pragma unroll
  for (int i = 0; i < 8; ++i) z[i] = 0.f;
  return z;
}

__device__ __forceinline__ v8f wmma_bf16(const Frag16& a, const Frag16& b, v8f c) {
  // v_wmma_f32_16x16x32_bf16: D = A(16x32) * B(32x16) + C
  return __builtin_amdgcn_wmma_f32_16x16x32_bf16(
      /*neg_a=*/false, a.v, /*neg_b=*/false, b.v,
      /*c_mod=*/(short)0, c, /*reuse_a=*/false, /*reuse_b=*/false);
}

// ---------------------------------------------------------------------------
// f32 -> bf16 conversion (4-wide)
// ---------------------------------------------------------------------------
__global__ void cvt4_kernel(const float* __restrict__ src, __bf16* __restrict__ dst,
                            size_t nvec) {
  size_t i = (size_t)blockIdx.x * blockDim.x + threadIdx.x;
  size_t stride = (size_t)gridDim.x * blockDim.x;
  for (; i < nvec; i += stride) {
    v4f x = *(const v4f*)(src + 4 * i);
    v4bf y;
#pragma unroll
    for (int j = 0; j < 4; ++j) y[j] = (__bf16)x[j];
    *(v4bf*)(dst + 4 * i) = y;
  }
}

// f32 (B,S,H) -> bf16 row-major (B,S,H) AND bf16 transposed (B,H,S)
__global__ __launch_bounds__(256)
void cvt_transpose_kernel(const float* __restrict__ src, __bf16* __restrict__ dstR,
                          __bf16* __restrict__ dstT, int S, int H) {
  __shared__ float tile[32][33];
  const int b  = blockIdx.z;
  const int s0 = blockIdx.y * 32;
  const int h0 = blockIdx.x * 32;
  const int tx = threadIdx.x;   // 0..31
  const int ty = threadIdx.y;   // 0..7
#pragma unroll
  for (int i = 0; i < 4; ++i) {
    int r = ty + i * 8;
    float v = src[((size_t)b * S + s0 + r) * H + h0 + tx];
    tile[r][tx] = v;
    dstR[((size_t)b * S + s0 + r) * H + h0 + tx] = (__bf16)v;
  }
  __syncthreads();
#pragma unroll
  for (int i = 0; i < 4; ++i) {
    int r = ty + i * 8;  // local h index
    dstT[((size_t)b * H + h0 + r) * S + s0 + tx] = (__bf16)tile[tx][r];
  }
}

// ---------------------------------------------------------------------------
// Fused attention: logits -> softmax (weights to d_out) -> ctx (bf16 to ws)
// One block = 16 query rows.  8 waves; wave w owns s-strip [w*NT*16, ...).
// ---------------------------------------------------------------------------
template <int S, int NT>
__global__ __launch_bounds__(256)
void attn_kernel(const __bf16* __restrict__ Q,    // (B,T,H) bf16
                 const __bf16* __restrict__ E,    // (B,S,H) bf16
                 const __bf16* __restrict__ ET,   // (B,H,S) bf16
                 float*        __restrict__ attw, // (B,T,S) f32 out
                 __bf16*       __restrict__ ctx)  // (B,T,H) bf16 out
{
  constexpr int T  = 1024, H = 1024;
  constexpr int SP = S + 8;                  // padded LDS row (bank-skew)
  __shared__ float red[8][16];
  __shared__ __align__(16) __bf16 attl[16 * SP];

  const int b     = blockIdx.x >> 6;         // T/16 == 64
  const int tbase = (blockIdx.x & 63) * 16;
  const int wave  = threadIdx.x >> 5;
  const int lane  = threadIdx.x & 31;
  const int n     = lane & 15;               // A row / B-C-D column index
  const int hi    = lane >> 4;               // half-wave selector

  // ---- Phase A: logits(16 x NT*16) = Q_tile(16xH) @ E_strip^T ----
  v8f acc[NT];
#pragma unroll
  for (int t = 0; t < NT; ++t) acc[t] = zero8();

  const __bf16* qrow  = Q + ((size_t)b * T + tbase + n) * H;
  const int     sbase = wave * (NT * 16);
  for (int k = 0; k < H; k += 32) {
    Frag16 a;  // A layout: lanes 0-15: K k..k+15 ; lanes 16-31: K k+16..k+31
    a.h[0] = *(const v8bf*)(qrow + k + hi * 16);
    a.h[1] = *(const v8bf*)(qrow + k + hi * 16 + 8);
#pragma unroll
    for (int t = 0; t < NT; ++t) {
      // B layout: lane n = column s; lanes 0-15 K k..k+15, lanes 16-31 K k+16..k+31
      const __bf16* erow = E + ((size_t)b * S + sbase + t * 16 + n) * H + k + hi * 16;
      Frag16 bb;
      bb.h[0] = *(const v8bf*)(erow);
      bb.h[1] = *(const v8bf*)(erow + 8);
      acc[t] = wmma_bf16(a, bb, acc[t]);
    }
  }

  // ---- Softmax over full S (cross-wave via LDS) ----
  // C layout: acc[t][j] = element (row j + 8*hi, col n) of tile t.
  float rmax[8];
#pragma unroll
  for (int j = 0; j < 8; ++j) {
    float v = acc[0][j];
#pragma unroll
    for (int t = 1; t < NT; ++t) v = fmaxf(v, acc[t][j]);
#pragma unroll
    for (int off = 1; off < 16; off <<= 1) v = fmaxf(v, __shfl_xor(v, off, 32));
    rmax[j] = v;
  }
  if (n == 0) {
#pragma unroll
    for (int j = 0; j < 8; ++j) red[wave][j + hi * 8] = rmax[j];
  }
  __syncthreads();
#pragma unroll
  for (int j = 0; j < 8; ++j) {
    float v = red[0][j + hi * 8];
#pragma unroll
    for (int w = 1; w < 8; ++w) v = fmaxf(v, red[w][j + hi * 8]);
    rmax[j] = v;
  }
  __syncthreads();

  float rsum[8];
#pragma unroll
  for (int j = 0; j < 8; ++j) {
    float s = 0.f;
#pragma unroll
    for (int t = 0; t < NT; ++t) {
      float e = __expf(acc[t][j] - rmax[j]);
      acc[t][j] = e;
      s += e;
    }
#pragma unroll
    for (int off = 1; off < 16; off <<= 1) s += __shfl_xor(s, off, 32);
    rsum[j] = s;
  }
  if (n == 0) {
#pragma unroll
    for (int j = 0; j < 8; ++j) red[wave][j + hi * 8] = rsum[j];
  }
  __syncthreads();
#pragma unroll
  for (int j = 0; j < 8; ++j) {
    float s = 0.f;
#pragma unroll
    for (int w = 0; w < 8; ++w) s += red[w][j + hi * 8];
    rsum[j] = 1.f / s;
  }

  // normalize; emit f32 weights (output) + bf16 att into LDS for phase B
#pragma unroll
  for (int t = 0; t < NT; ++t) {
#pragma unroll
    for (int j = 0; j < 8; ++j) {
      float p = acc[t][j] * rsum[j];
      int m    = j + hi * 8;
      int scol = sbase + t * 16 + n;
      attw[((size_t)b * T + tbase + m) * S + scol] = p;
      attl[m * SP + scol] = (__bf16)p;
    }
  }
  __syncthreads();

  // ---- Phase B: ctx(16x128 per wave) = att(16xS) @ E(SxH) via E^T rows ----
  v8f cacc[8];
#pragma unroll
  for (int t = 0; t < 8; ++t) cacc[t] = zero8();
  const __bf16* arow = attl + n * SP;
  for (int k = 0; k < S; k += 32) {
    Frag16 a;
    a.h[0] = *(const v8bf*)(arow + k + hi * 16);
    a.h[1] = *(const v8bf*)(arow + k + hi * 16 + 8);
#pragma unroll
    for (int t = 0; t < 8; ++t) {
      const __bf16* etrow =
          ET + ((size_t)b * H + wave * 128 + t * 16 + n) * S + k + hi * 16;
      Frag16 bb;
      bb.h[0] = *(const v8bf*)(etrow);
      bb.h[1] = *(const v8bf*)(etrow + 8);
      cacc[t] = wmma_bf16(a, bb, cacc[t]);
    }
  }
#pragma unroll
  for (int t = 0; t < 8; ++t) {
#pragma unroll
    for (int j = 0; j < 8; ++j) {
      int m = j + hi * 8;
      ctx[((size_t)b * T + tbase + m) * H + wave * 128 + t * 16 + n] =
          (__bf16)cacc[t][j];
    }
  }
}

// ---------------------------------------------------------------------------
// Gate + fusion:  gate = sigmoid([ctx1,ctx2,out] @ Wg^T)
//                 fusion = tanh((1-g)*ctx1@Ws^T + g*ctx2@Wt^T + out@Wo^T)
// grid.y tiles H in chunks of 256; each wave owns 2 16x16 output tiles.
// ---------------------------------------------------------------------------
__global__ __launch_bounds__(256)
void fusion_kernel(const __bf16* __restrict__ C1, const __bf16* __restrict__ C2,
                   const __bf16* __restrict__ Q,  const __bf16* __restrict__ Wg,
                   const __bf16* __restrict__ Ws, const __bf16* __restrict__ Wt,
                   const __bf16* __restrict__ Wo, float* __restrict__ out) {
  constexpr int T = 1024, H = 1024;
  const int b     = blockIdx.x >> 6;
  const int tbase = (blockIdx.x & 63) * 16;
  const int wave  = threadIdx.x >> 5;
  const int lane  = threadIdx.x & 31;
  const int n     = lane & 15;
  const int hi    = lane >> 4;
  const int hbase = blockIdx.y * 256 + wave * 32;

  v8f gacc[2], sacc[2], tacc[2], oacc[2];
#pragma unroll
  for (int t = 0; t < 2; ++t) {
    gacc[t] = zero8(); sacc[t] = zero8(); tacc[t] = zero8(); oacc[t] = zero8();
  }

  const size_t rowoff = ((size_t)b * T + tbase + n) * H;
  const __bf16* c1r = C1 + rowoff;
  const __bf16* c2r = C2 + rowoff;
  const __bf16* qr  = Q  + rowoff;

  for (int k = 0; k < H; k += 32) {
    Frag16 a1, a2, a3;
    a1.h[0] = *(const v8bf*)(c1r + k + hi * 16);
    a1.h[1] = *(const v8bf*)(c1r + k + hi * 16 + 8);
    a2.h[0] = *(const v8bf*)(c2r + k + hi * 16);
    a2.h[1] = *(const v8bf*)(c2r + k + hi * 16 + 8);
    a3.h[0] = *(const v8bf*)(qr + k + hi * 16);
    a3.h[1] = *(const v8bf*)(qr + k + hi * 16 + 8);
#pragma unroll
    for (int t = 0; t < 2; ++t) {
      const int ncol = hbase + t * 16 + n;
      const __bf16* wgr = Wg + (size_t)ncol * (3 * H) + k + hi * 16;
      Frag16 bb;
      // gate: cat = [ctx1 | ctx2 | out] against Wg row segments
      bb.h[0] = *(const v8bf*)(wgr);             bb.h[1] = *(const v8bf*)(wgr + 8);
      gacc[t] = wmma_bf16(a1, bb, gacc[t]);
      bb.h[0] = *(const v8bf*)(wgr + H);         bb.h[1] = *(const v8bf*)(wgr + H + 8);
      gacc[t] = wmma_bf16(a2, bb, gacc[t]);
      bb.h[0] = *(const v8bf*)(wgr + 2 * H);     bb.h[1] = *(const v8bf*)(wgr + 2 * H + 8);
      gacc[t] = wmma_bf16(a3, bb, gacc[t]);
      // projections
      const __bf16* wsr = Ws + (size_t)ncol * H + k + hi * 16;
      bb.h[0] = *(const v8bf*)(wsr);             bb.h[1] = *(const v8bf*)(wsr + 8);
      sacc[t] = wmma_bf16(a1, bb, sacc[t]);
      const __bf16* wtr = Wt + (size_t)ncol * H + k + hi * 16;
      bb.h[0] = *(const v8bf*)(wtr);             bb.h[1] = *(const v8bf*)(wtr + 8);
      tacc[t] = wmma_bf16(a2, bb, tacc[t]);
      const __bf16* wor = Wo + (size_t)ncol * H + k + hi * 16;
      bb.h[0] = *(const v8bf*)(wor);             bb.h[1] = *(const v8bf*)(wor + 8);
      oacc[t] = wmma_bf16(a3, bb, oacc[t]);
    }
  }

#pragma unroll
  for (int t = 0; t < 2; ++t) {
#pragma unroll
    for (int j = 0; j < 8; ++j) {
      float g = 1.f / (1.f + __expf(-gacc[t][j]));
      float f = tanhf((1.f - g) * sacc[t][j] + g * tacc[t][j] + oacc[t][j]);
      int m = j + hi * 8;
      out[((size_t)b * T + tbase + m) * H + hbase + t * 16 + n] = f;
    }
  }
}

// ---------------------------------------------------------------------------
// Host-side orchestration
// ---------------------------------------------------------------------------
extern "C" void kernel_launch(void* const* d_in, const int* in_sizes, int n_in,
                              void* d_out, int out_size, void* d_ws, size_t ws_size,
                              hipStream_t stream) {
  (void)in_sizes; (void)n_in; (void)out_size; (void)ws_size;
  constexpr int B = 8, T = 1024, S1 = 1024, S2 = 512, H = 1024;

  const float* f_out  = (const float*)d_in[0];  // output  (B,T,H)
  const float* f_sent = (const float*)d_in[1];  // sent    (B,S1,H)
  const float* f_tmpl = (const float*)d_in[2];  // template(B,S2,H)
  const float* f_Wg   = (const float*)d_in[3];  // (H,3H)
  const float* f_Ws   = (const float*)d_in[4];  // (H,H)
  const float* f_Wt   = (const float*)d_in[5];  // (H,H)
  const float* f_Wo   = (const float*)d_in[6];  // (H,H)

  __bf16* ws = (__bf16*)d_ws;
  size_t o = 0;
  auto alloc = [&](size_t nelem) { __bf16* p = ws + o; o += nelem; return p; };
  __bf16* Qbf = alloc((size_t)B * T * H);
  __bf16* E1  = alloc((size_t)B * S1 * H);
  __bf16* E1T = alloc((size_t)B * H * S1);
  __bf16* E2  = alloc((size_t)B * S2 * H);
  __bf16* E2T = alloc((size_t)B * H * S2);
  __bf16* Wg  = alloc((size_t)H * 3 * H);
  __bf16* Wsb = alloc((size_t)H * H);
  __bf16* Wtb = alloc((size_t)H * H);
  __bf16* Wob = alloc((size_t)H * H);
  __bf16* C1  = alloc((size_t)B * T * H);
  __bf16* C2  = alloc((size_t)B * T * H);

  float* fusion = (float*)d_out;                       // (B,T,H)
  float* sentw  = fusion + (size_t)B * T * H;          // (B,T,S1)
  float* tmplw  = sentw  + (size_t)B * T * S1;         // (B,T,S2)

  // bf16 conversions
  cvt4_kernel<<<2048, 256, 0, stream>>>(f_out, Qbf, (size_t)B * T * H / 4);
  cvt4_kernel<<<1024, 256, 0, stream>>>(f_Wg, Wg, (size_t)H * 3 * H / 4);
  cvt4_kernel<<<512, 256, 0, stream>>>(f_Ws, Wsb, (size_t)H * H / 4);
  cvt4_kernel<<<512, 256, 0, stream>>>(f_Wt, Wtb, (size_t)H * H / 4);
  cvt4_kernel<<<512, 256, 0, stream>>>(f_Wo, Wob, (size_t)H * H / 4);
  cvt_transpose_kernel<<<dim3(H / 32, S1 / 32, B), dim3(32, 8), 0, stream>>>(
      f_sent, E1, E1T, S1, H);
  cvt_transpose_kernel<<<dim3(H / 32, S2 / 32, B), dim3(32, 8), 0, stream>>>(
      f_tmpl, E2, E2T, S2, H);

  // attention (sent) : 8 tiles/wave over S=1024
  attn_kernel<S1, 8><<<B * (T / 16), 256, 0, stream>>>(Qbf, E1, E1T, sentw, C1);
  // attention (template) : 4 tiles/wave over S=512
  attn_kernel<S2, 4><<<B * (T / 16), 256, 0, stream>>>(Qbf, E2, E2T, tmplw, C2);

  // gate + fusion
  fusion_kernel<<<dim3(B * (T / 16), H / 256), 256, 0, stream>>>(
      C1, C2, Qbf, Wg, Wsb, Wtb, Wob, fusion);
}
